// TripletLoss_65489661329590
// MI455X (gfx1250) — matrix operand order repaired and checked by reading, not compiled, not run on hardware
//
#include <hip/hip_runtime.h>
#include <cstdint>

typedef __bf16 v16bf __attribute__((ext_vector_type(16)));
typedef __bf16 v8bf __attribute__((ext_vector_type(8)));
typedef __bf16 v4bf __attribute__((ext_vector_type(4)));
typedef float v8f __attribute__((ext_vector_type(8)));
typedef float v4f __attribute__((ext_vector_type(4)));

#define DIM 128
#define BROW 136  // padded LDS row stride (bf16 elems): 272 B = +4 banks/row
#define MARGIN 0.5f

__device__ __forceinline__ uint32_t map_ord(float f) {
  uint32_t u = __float_as_uint(f);
  return (u & 0x80000000u) ? ~u : (u | 0x80000000u);
}
__device__ __forceinline__ float unmap_ord(uint32_t u) {
  return (u & 0x80000000u) ? __uint_as_float(u ^ 0x80000000u)
                           : __uint_as_float(~u);
}

__global__ void tl_init_kernel(uint32_t* __restrict__ pos_u,
                               uint32_t* __restrict__ neg_u, int n) {
  int i = threadIdx.x + blockIdx.x * blockDim.x;
  if (i < n) {
    pos_u[i] = 0xFFFFFFFFu;  // mapped +inf (min identity)
    neg_u[i] = 0x00000000u;  // mapped -inf (max identity)
  }
}

// B fragment: 16 consecutive bf16 from LDS as two 16B-aligned 8-elt loads.
__device__ __forceinline__ v16bf load_bfrag(const __bf16* p) {
  v8bf a = *(const v8bf*)p;
  v8bf b = *(const v8bf*)(p + 8);
  return __builtin_shufflevector(a, b, 0, 1, 2, 3, 4, 5, 6, 7, 8, 9, 10, 11,
                                 12, 13, 14, 15);
}

// Fused sim = inputs @ features^T with hardest-pos(min)/hardest-neg(max).
// Two-term bf16 split: sim = Ah*Bh + Al*Bh + Ah*Bl (f32 accumulate) on the
// bf16 16x16x32 WMMA pipe; features read from HBM exactly once and staged
// as bf16 hi/lo in padded (conflict-free) LDS.
__global__ __launch_bounds__(256) void tl_sim_kernel(
    const float* __restrict__ inputs,    // [256, 128]
    const float* __restrict__ features,  // [n_feat, 128]
    const int* __restrict__ targets,     // [256]
    const int* __restrict__ labels,      // [n_feat]
    const int* __restrict__ idxv,        // [256]
    uint32_t* __restrict__ pos_u,        // [256] mapped
    uint32_t* __restrict__ neg_u,        // [256] mapped
    int n_feat) {
  __shared__ __align__(16) __bf16 sBh[128 * BROW];  // 34 KB
  __shared__ __align__(16) __bf16 sBl[128 * BROW];  // 34 KB

  const int tid  = threadIdx.x;
  const int wave = tid >> 5;
  const int lane = tid & 31;
  const int half = lane >> 4;
  const int ln   = lane & 15;
  const int col0 = blockIdx.x * 128;
  const int n_ct = min(8, (n_feat - col0) >> 4);  // n_feat % 16 == 0

  // ---- Stage features tile -> bf16 hi/lo in LDS (one HBM read). ----
  {
    const int c  = tid >> 1;
    const int k0 = (tid & 1) * 64;
    const int src = min(col0 + c, n_feat - 1);
    const float* srow = features + (size_t)src * DIM + k0;
    __bf16* dh = sBh + c * BROW + k0;
    __bf16* dl = sBl + c * BROW + k0;
#pragma unroll
    for (int k = 0; k < 64; k += 4) {
      v4f x = *(const v4f*)(srow + k);
      v4bf hv, lv;
#pragma unroll
      for (int u = 0; u < 4; ++u) {
        __bf16 h = (__bf16)x[u];
        hv[u] = h;
        lv[u] = (__bf16)(x[u] - (float)h);
      }
      *(v4bf*)(dh + k) = hv;
      *(v4bf*)(dl + k) = lv;
    }
  }
  __syncthreads();

  const int i_base = wave * 32;  // this wave's 32 batch rows (2 M-tiles)

  // ---- A fragments (hi+lo, both M-tiles, 4 K-steps) in registers. ----
  v16bf ahf[2][4], alf[2][4];
#pragma unroll
  for (int t2 = 0; t2 < 2; ++t2) {
    const float* arow = inputs + (size_t)(i_base + t2 * 16 + ln) * DIM;
#pragma unroll
    for (int s = 0; s < 4; ++s) {
#pragma unroll
      for (int e = 0; e < 16; ++e) {
        const int k = s * 32 + half * 16 + e;
        const float x = arow[k];
        const __bf16 h = (__bf16)x;
        ahf[t2][s][e] = h;
        alf[t2][s][e] = (__bf16)(x - (float)h);
      }
    }
  }

  int tgt[2][8], slf[2][8];
  float rmin[2][8], rmax[2][8];
#pragma unroll
  for (int t2 = 0; t2 < 2; ++t2)
#pragma unroll
    for (int r = 0; r < 8; ++r) {
      const int i = i_base + t2 * 16 + half * 8 + r;
      tgt[t2][r] = targets[i];
      slf[t2][r] = idxv[i];
      rmin[t2][r] = 3.4028234e38f;
      rmax[t2][r] = -3.4028234e38f;
    }

  for (int ct = 0; ct < n_ct; ++ct) {
    const int jl = ct * 16 + ln;
    const int j  = col0 + jl;
    const __bf16* bph = sBh + jl * BROW;
    const __bf16* bpl = sBl + jl * BROW;

    v8f acc[2] = {v8f{}, v8f{}};
#pragma unroll
    for (int s = 0; s < 4; ++s) {
      const v16bf bh = load_bfrag(bph + s * 32 + half * 16);
      const v16bf bl = load_bfrag(bpl + s * 32 + half * 16);
      acc[0] = __builtin_amdgcn_wmma_f32_16x16x32_bf16(
          false, ahf[0][s], false, bh, (short)0, acc[0], false, false);
      acc[1] = __builtin_amdgcn_wmma_f32_16x16x32_bf16(
          false, ahf[1][s], false, bh, (short)0, acc[1], false, false);
      acc[0] = __builtin_amdgcn_wmma_f32_16x16x32_bf16(
          false, alf[0][s], false, bh, (short)0, acc[0], false, false);
      acc[1] = __builtin_amdgcn_wmma_f32_16x16x32_bf16(
          false, alf[1][s], false, bh, (short)0, acc[1], false, false);
      acc[0] = __builtin_amdgcn_wmma_f32_16x16x32_bf16(
          false, ahf[0][s], false, bl, (short)0, acc[0], false, false);
      acc[1] = __builtin_amdgcn_wmma_f32_16x16x32_bf16(
          false, ahf[1][s], false, bl, (short)0, acc[1], false, false);
    }

    const int lab = labels[j];
#pragma unroll
    for (int t2 = 0; t2 < 2; ++t2) {
#pragma unroll
      for (int r = 0; r < 8; ++r) {
        const bool same = (lab == tgt[t2][r]);
        const bool self = (slf[t2][r] == j);
        const float v = acc[t2][r];
        if (same & !self & (v < rmin[t2][r])) rmin[t2][r] = v;
        if ((!same) & (v > rmax[t2][r])) rmax[t2][r] = v;
      }
    }
  }

#pragma unroll
  for (int off = 1; off < 16; off <<= 1) {
#pragma unroll
    for (int t2 = 0; t2 < 2; ++t2)
#pragma unroll
      for (int r = 0; r < 8; ++r) {
        const float omin = __shfl_xor(rmin[t2][r], off, 32);
        const float omax = __shfl_xor(rmax[t2][r], off, 32);
        if (omin < rmin[t2][r]) rmin[t2][r] = omin;
        if (omax > rmax[t2][r]) rmax[t2][r] = omax;
      }
  }
  if (ln == 0) {
#pragma unroll
    for (int t2 = 0; t2 < 2; ++t2)
#pragma unroll
      for (int r = 0; r < 8; ++r) {
        const int i = i_base + t2 * 16 + half * 8 + r;
        atomicMin(pos_u + i, map_ord(rmin[t2][r]));
        atomicMax(neg_u + i, map_ord(rmax[t2][r]));
      }
  }
}

__global__ __launch_bounds__(256) void tl_finalize_kernel(
    const uint32_t* __restrict__ pos_u, const uint32_t* __restrict__ neg_u,
    float* __restrict__ out, int n) {
  __shared__ float sm[256];
  const int i = threadIdx.x;
  const float pos = unmap_ord(pos_u[i]);
  const float neg = unmap_ord(neg_u[i]);
  sm[i] = fmaxf(neg - pos + MARGIN, 0.0f);
  __syncthreads();
  for (int s = 128; s > 0; s >>= 1) {
    if (i < s) sm[i] += sm[i + s];
    __syncthreads();
  }
  if (i == 0) out[0] = sm[0] / (float)n;
}

extern "C" void kernel_launch(void* const* d_in, const int* in_sizes, int n_in,
                              void* d_out, int out_size, void* d_ws,
                              size_t ws_size, hipStream_t stream) {
  const float* inputs   = (const float*)d_in[0];
  const float* features = (const float*)d_in[1];
  const int* targets    = (const int*)d_in[2];
  const int* labels     = (const int*)d_in[3];
  const int* idxv       = (const int*)d_in[4];
  const int n_batch = in_sizes[2];  // 256
  const int n_feat  = in_sizes[3];  // 100000

  uint32_t* pos_u = (uint32_t*)d_ws;
  uint32_t* neg_u = pos_u + n_batch;

  tl_init_kernel<<<1, 256, 0, stream>>>(pos_u, neg_u, n_batch);

  const int nblk = (n_feat + 127) / 128;
  tl_sim_kernel<<<nblk, 256, 0, stream>>>(inputs, features, targets, labels,
                                          idxv, pos_u, neg_u, n_feat);

  tl_finalize_kernel<<<1, n_batch, 0, stream>>>(pos_u, neg_u, (float*)d_out,
                                                n_batch);
}
